// Transformer_76991583748322
// MI455X (gfx1250) — matrix operand order repaired and checked by reading, not compiled, not run on hardware
//
#include <hip/hip_runtime.h>
#include <hip/hip_bf16.h>

// ---------------- shapes (compile-time constants from reference) ----------------
#define LEN     7681        // 76*76 + 38*38 + 19*19 + 10*10
#define DM      256
#define NH      8
#define DH      32
#define NLVL    4
#define NPT     4
#define DFF     1024
#define NLAYERS 6

typedef __attribute__((ext_vector_type(16))) _Float16 v16h;
typedef __attribute__((ext_vector_type(8)))  _Float16 v8h;
typedef __attribute__((ext_vector_type(8)))  float    v8f;

// ---------------- elementwise: q16 = (f16)(a + b) ----------------
__global__ void k_add_h(const float* __restrict__ a, const float* __restrict__ b,
                        _Float16* __restrict__ o, int n) {
    int i = blockIdx.x * 256 + threadIdx.x;
    if (i < n) o[i] = (_Float16)(a[i] + b[i]);
}

// ---------------- f32 -> f16 copy ----------------
__global__ void k_cvt(const float* __restrict__ a, _Float16* __restrict__ o, int n) {
    int i = blockIdx.x * 256 + threadIdx.x;
    if (i < n) o[i] = (_Float16)a[i];
}

// ---------------- weight transpose + f32->f16:  WT[n*K+k] = W[k*N+n] ----------------
__global__ void k_wtrans(const float* __restrict__ W, _Float16* __restrict__ WT,
                         int K, int N) {
    int i = blockIdx.x * 256 + threadIdx.x;
    if (i >= K * N) return;
    int n = i / K;
    int k = i - n * K;
    WT[i] = (_Float16)W[(size_t)k * N + n];
}

// ---------------- WMMA fragment loaders (wave32, 16x16x32 f16) ----------------
// A 16x32 f16: lanes 0-15 -> M=lane,    K = {k0..k0+15}
//              lanes 16-31 -> M=lane-16, K = {k0+16..k0+31}
__device__ __forceinline__ v16h frag_a(const _Float16* __restrict__ p /* row base+half */,
                                       int k0) {
    v8h lo = *(const v8h*)(p + k0);
    v8h hi = *(const v8h*)(p + k0 + 8);
    return __builtin_shufflevector(lo, hi, 0,1,2,3,4,5,6,7,8,9,10,11,12,13,14,15);
}

// ---------------- WMMA GEMM: C[M,N] = A[M,K] @ W[K,N] + bias ----------------
// Each wave computes a 32x64 register tile: 2 A fragments x 4 B fragments ->
// 8 WMMAs per k-step on 12 b128 loads. The 4 waves of a block cover 8 consecutive
// M-tiles at the same N-strip so their B fragment addresses coincide (WGP$ hits).
// WT is N-major f16.
template<int RELU, int OUTF16>
__global__ void k_gemm(const _Float16* __restrict__ A, const _Float16* __restrict__ WT,
                       const float* __restrict__ bias, float* __restrict__ Cf,
                       _Float16* __restrict__ Ch, int M, int N, int K) {
    int lane    = threadIdx.x & 31;
    int wid     = threadIdx.x >> 5;
    int nchunks = N >> 6;                     // 64-wide N strips
    int g8      = blockIdx.x / nchunks;       // group of 8 M-tiles
    int nc      = blockIdx.x - g8 * nchunks;
    int mt0     = g8 * 8 + wid * 2;           // this wave: M-tiles mt0, mt0+1
    int mtiles  = (M + 15) >> 4;
    if (mt0 >= mtiles) return;

    int lr    = lane & 15;
    int ahalf = (lane & 16) ? 16 : 0;
    int row0  = mt0 * 16 + lr;       if (row0 > M - 1) row0 = M - 1;  // clamp loads
    int row1  = (mt0 + 1) * 16 + lr; if (row1 > M - 1) row1 = M - 1;
    int n0    = nc * 64 + lr;

    const _Float16* ap0 = A  + (size_t)row0 * K + ahalf;
    const _Float16* ap1 = A  + (size_t)row1 * K + ahalf;
    const _Float16* bp  = WT + (size_t)n0   * K + ((lane & 16) ? 16 : 0);
    const size_t    bs  = (size_t)16 * K;     // stride between the 4 B tiles

    v8f z = {0.f,0.f,0.f,0.f,0.f,0.f,0.f,0.f};
    v8f c00 = z, c01 = z, c02 = z, c03 = z;   // M-tile 0 x N-tiles 0..3
    v8f c10 = z, c11 = z, c12 = z, c13 = z;   // M-tile 1 x N-tiles 0..3
    for (int k0 = 0; k0 < K; k0 += 32) {
        v16h a0 = frag_a(ap0, k0);
        v16h a1 = frag_a(ap1, k0);
        v16h b0 = *(const v16h*)(bp + k0);
        v16h b1 = *(const v16h*)(bp + bs     + k0);
        v16h b2 = *(const v16h*)(bp + bs * 2 + k0);
        v16h b3 = *(const v16h*)(bp + bs * 3 + k0);
        c00 = __builtin_amdgcn_wmma_f32_16x16x32_f16(false, a0, false, b0, (short)0, c00, false, false);
        c01 = __builtin_amdgcn_wmma_f32_16x16x32_f16(false, a0, false, b1, (short)0, c01, false, false);
        c02 = __builtin_amdgcn_wmma_f32_16x16x32_f16(false, a0, false, b2, (short)0, c02, false, false);
        c03 = __builtin_amdgcn_wmma_f32_16x16x32_f16(false, a0, false, b3, (short)0, c03, false, false);
        c10 = __builtin_amdgcn_wmma_f32_16x16x32_f16(false, a1, false, b0, (short)0, c10, false, false);
        c11 = __builtin_amdgcn_wmma_f32_16x16x32_f16(false, a1, false, b1, (short)0, c11, false, false);
        c12 = __builtin_amdgcn_wmma_f32_16x16x32_f16(false, a1, false, b2, (short)0, c12, false, false);
        c13 = __builtin_amdgcn_wmma_f32_16x16x32_f16(false, a1, false, b3, (short)0, c13, false, false);
    }

    v8f accs[2][4] = {{c00, c01, c02, c03}, {c10, c11, c12, c13}};
#pragma unroll
    for (int p = 0; p < 2; p++) {
        int mbase = (mt0 + p) * 16 + ((lane & 16) ? 8 : 0);
#pragma unroll
        for (int t = 0; t < 4; t++) {
            int   n  = n0 + t * 16;
            float bn = bias[n];
#pragma unroll
            for (int r = 0; r < 8; r++) {
                int m = mbase + r;
                if (m < M) {
                    float v = accs[p][t][r] + bn;
                    if (RELU) v = fmaxf(v, 0.0f);
                    if (OUTF16) Ch[(size_t)m * N + n] = (_Float16)v;
                    else        Cf[(size_t)m * N + n] = v;
                }
            }
        }
    }
}

// ---------------- softmax over 16 attention logits per (q, head) ----------------
__global__ void k_softmax16(float* __restrict__ aw, int n) {   // n = LEN*NH
    int i = blockIdx.x * 256 + threadIdx.x;
    if (i >= n) return;
    float* p = aw + (size_t)i * 16;
    float mx = p[0];
#pragma unroll
    for (int j = 1; j < 16; j++) mx = fmaxf(mx, p[j]);
    float e[16], s = 0.f;
#pragma unroll
    for (int j = 0; j < 16; j++) { e[j] = __expf(p[j] - mx); s += e[j]; }
    float inv = 1.f / s;
#pragma unroll
    for (int j = 0; j < 16; j++) p[j] = e[j] * inv;
}

// ---------------- deformable-attention sampling (f16 output for oproj GEMM) ------
// One wave per (q, head); lane = channel d (DH==32). Each bilinear corner fetch is
// one coalesced 128B load across the 32 head-dims; address math is lane-uniform.
__global__ void k_msda(const float* __restrict__ value, const float* __restrict__ off,
                       const float* __restrict__ aw, _Float16* __restrict__ out) {
    int wv   = blockIdx.x * 8 + (threadIdx.x >> 5);
    int lane = threadIdx.x & 31;
    int q = wv >> 3, h = wv & 7;
    if (q >= LEN) return;

    constexpr int HS[4] = {76, 38, 19, 10};
    constexpr int WS[4] = {76, 38, 19, 10};
    constexpr int ST[4] = {0, 5776, 7220, 7581};

    // reference point of query q (valid_ratios == 1)
    int lvq, loc;
    if      (q < 5776) { lvq = 0; loc = q; }
    else if (q < 7220) { lvq = 1; loc = q - 5776; }
    else if (q < 7581) { lvq = 2; loc = q - 7220; }
    else               { lvq = 3; loc = q - 7581; }
    int   Wq = WS[lvq], rI = loc / Wq, cI = loc - rI * Wq;
    float rx = (cI + 0.5f) / (float)Wq;
    float ry = (rI + 0.5f) / (float)HS[lvq];

    const float* offq = off + (size_t)q * (NH * NLVL * NPT * 2) + h * (NLVL * NPT * 2);
    const float* awq  = aw  + (size_t)q * (NH * NLVL * NPT)     + h * (NLVL * NPT);

    float acc = 0.f;
#pragma unroll
    for (int lv = 0; lv < NLVL; lv++) {
        const int Hh = HS[lv], Ww = WS[lv];
        const float* vl = value + (size_t)ST[lv] * DM + h * DH;
#pragma unroll
        for (int pt = 0; pt < NPT; pt++) {
            float ox = offq[(lv * NPT + pt) * 2 + 0];
            float oy = offq[(lv * NPT + pt) * 2 + 1];
            float w  = awq[lv * NPT + pt];
            float x = rx * (float)Ww + ox - 0.5f;
            float y = ry * (float)Hh + oy - 0.5f;
            float x0f = floorf(x), y0f = floorf(y);
            float wx = x - x0f, wy = y - y0f;
            int x0 = (int)x0f, y0 = (int)y0f;

            auto g = [&](int xi, int yi) -> float {
                bool valid = (xi >= 0) & (xi < Ww) & (yi >= 0) & (yi < Hh);
                int xc = min(max(xi, 0), Ww - 1);
                int yc = min(max(yi, 0), Hh - 1);
                float v = vl[(size_t)(yc * Ww + xc) * DM + lane];
                return valid ? v : 0.f;
            };
            float s = g(x0,     y0    ) * (1.f - wx) * (1.f - wy)
                    + g(x0 + 1, y0    ) * wx         * (1.f - wy)
                    + g(x0,     y0 + 1) * (1.f - wx) * wy
                    + g(x0 + 1, y0 + 1) * wx         * wy;
            acc += w * s;
        }
    }
    out[(size_t)q * DM + h * DH + lane] = (_Float16)acc;
}

// ------- fused residual-add + LayerNorm (one wave per row), f32 + f16 outputs -----
__global__ void k_add_ln(const float* __restrict__ a, const float* __restrict__ b,
                         const float* __restrict__ g, const float* __restrict__ be,
                         float* __restrict__ out, _Float16* __restrict__ out16,
                         float* __restrict__ out2) {
    int w    = blockIdx.x * 8 + (threadIdx.x >> 5);
    int lane = threadIdx.x & 31;
    if (w >= LEN) return;
    const float* pa = a + (size_t)w * DM;
    const float* pb = b + (size_t)w * DM;
    float x[8], s = 0.f;
#pragma unroll
    for (int j = 0; j < 8; j++) { x[j] = pa[lane + 32 * j] + pb[lane + 32 * j]; s += x[j]; }
#pragma unroll
    for (int o = 16; o >= 1; o >>= 1) s += __shfl_xor(s, o, 32);
    float mean = s * (1.f / DM);
    float v = 0.f;
#pragma unroll
    for (int j = 0; j < 8; j++) { float d = x[j] - mean; v += d * d; }
#pragma unroll
    for (int o = 16; o >= 1; o >>= 1) v += __shfl_xor(v, o, 32);
    float rstd = rsqrtf(v * (1.f / DM) + 1e-5f);
    float*     po  = out   + (size_t)w * DM;
    _Float16*  ph  = out16 + (size_t)w * DM;
    float*     po2 = out2 ? out2 + (size_t)w * DM : nullptr;
#pragma unroll
    for (int j = 0; j < 8; j++) {
        int c = lane + 32 * j;
        float y = (x[j] - mean) * rstd * g[c] + be[c];
        po[c] = y;
        ph[c] = (_Float16)y;
        if (out2) po2[c] = y;
    }
}

// ---------------- host orchestration ----------------
extern "C" void kernel_launch(void* const* d_in, const int* in_sizes, int n_in,
                              void* d_out, int out_size, void* d_ws, size_t ws_size,
                              hipStream_t stream) {
    // Top-level inputs in setup_inputs() dict order; 'params' pytree flattened in
    // jax order (dict keys sorted alphabetically, (W,b) tuples in order).
    const float* src = (const float*)d_in[0];
    // d_in[1] spatial_shapes, d_in[2] level_start_index, d_in[3] valid_ratios: constants
    const float* pos = (const float*)d_in[4];
    const float* aw_w  = (const float*)d_in[5];
    const float* aw_b  = (const float*)d_in[6];
    const float* fc1_w = (const float*)d_in[7];
    const float* fc1_b = (const float*)d_in[8];
    const float* fc2_w = (const float*)d_in[9];
    const float* fc2_b = (const float*)d_in[10];
    const float* ln1_g = (const float*)d_in[11];
    const float* ln1_b = (const float*)d_in[12];
    const float* ln2_g = (const float*)d_in[13];
    const float* ln2_b = (const float*)d_in[14];
    const float* off_w = (const float*)d_in[15];
    const float* off_b = (const float*)d_in[16];
    const float* op_w  = (const float*)d_in[17];
    const float* op_b  = (const float*)d_in[18];
    const float* vp_w  = (const float*)d_in[19];
    const float* vp_b  = (const float*)d_in[20];
    float* out = (float*)d_out;

    // workspace carve-up (all chunks multiples of 256 B)
    char* w = (char*)d_ws;
    _Float16* q16    = (_Float16*)w; w += (size_t)LEN * DM  * 2;   // q / ln1 out (fc1 in)
    _Float16* cur16  = (_Float16*)w; w += (size_t)LEN * DM  * 2;   // f16 shadow of layer input
    _Float16* atn16  = (_Float16*)w; w += (size_t)LEN * DM  * 2;   // sampled (oproj in)
    float*    offbuf = (float*)w;    w += (size_t)LEN * DM  * 4;   // offsets / src2
    float*    awbuf  = (float*)w;    w += (size_t)LEN * 128 * 4;
    float*    valbuf = (float*)w;    w += (size_t)LEN * DM  * 4;   // value / ffn out
    float*    curbuf = (float*)w;    w += (size_t)LEN * DM  * 4;   // residual stream (f32)
    _Float16* mid16  = (_Float16*)w; w += (size_t)LEN * DFF * 2;   // relu(fc1) f16
    _Float16* wt     = (_Float16*)w; w += (size_t)DFF * DM  * 2;   // transposed weight

    const int EW_BLOCKS = (LEN * DM + 255) / 256;          // 7681
    const int LN_BLOCKS = (LEN + 7) / 8;                   // 961
    const int SM_BLOCKS = (LEN * NH + 255) / 256;          // 241
    const int MS_BLOCKS = (LEN * NH + 7) / 8;              // 7681
    const int MT8       = ((LEN + 15) / 16 + 7) / 8;       // groups of 8 M-tiles

    auto gemm = [&](const _Float16* A, const float* Wsrc, const float* bsrc,
                    int K, int N, float* Cf, _Float16* Ch, int relu) {
        int kn = K * N;
        k_wtrans<<<(kn + 255) / 256, 256, 0, stream>>>(Wsrc, wt, K, N);
        int blocks = MT8 * (N / 64);
        if (relu) k_gemm<1,1><<<blocks, 128, 0, stream>>>(A, wt, bsrc, nullptr, Ch, LEN, N, K);
        else      k_gemm<0,0><<<blocks, 128, 0, stream>>>(A, wt, bsrc, Cf, nullptr, LEN, N, K);
    };

    // f16 shadow of the initial src (layer-0 vproj input)
    k_cvt<<<EW_BLOCKS, 256, 0, stream>>>(src, cur16, LEN * DM);

    for (int l = 0; l < NLAYERS; l++) {
        const float* cur = (l == 0) ? src : curbuf;

        // q = src + pos  (f16 for the GEMMs)
        k_add_h<<<EW_BLOCKS, 256, 0, stream>>>(cur, pos, q16, LEN * DM);

        // sampling offsets, attention logits, value projection
        gemm(q16,   off_w + (size_t)l * DM * DM,  off_b + (size_t)l * DM,  DM, DM,  offbuf, nullptr, 0);
        gemm(q16,   aw_w  + (size_t)l * DM * 128, aw_b  + (size_t)l * 128, DM, 128, awbuf,  nullptr, 0);
        k_softmax16<<<SM_BLOCKS, 256, 0, stream>>>(awbuf, LEN * NH);
        gemm(cur16, vp_w  + (size_t)l * DM * DM,  vp_b  + (size_t)l * DM,  DM, DM,  valbuf, nullptr, 0);

        // bilinear sampling + attention-weighted sum -> f16
        k_msda<<<MS_BLOCKS, 256, 0, stream>>>(valbuf, offbuf, awbuf, atn16);

        // output projection (src2 -> offbuf), then LN1 -> curbuf(f32) + q16(f16)
        gemm(atn16, op_w + (size_t)l * DM * DM, op_b + (size_t)l * DM, DM, DM, offbuf, nullptr, 0);
        k_add_ln<<<LN_BLOCKS, 256, 0, stream>>>(cur, offbuf,
                                                ln1_g + (size_t)l * DM, ln1_b + (size_t)l * DM,
                                                curbuf, q16, nullptr);

        // FFN: fc1 (relu, f16 out) then fc2 -> valbuf
        gemm(q16,   fc1_w + (size_t)l * DM * DFF, fc1_b + (size_t)l * DFF, DM,  DFF, nullptr, mid16, 1);
        gemm(mid16, fc2_w + (size_t)l * DFF * DM, fc2_b + (size_t)l * DM,  DFF, DM,  valbuf, nullptr, 0);

        // LN2 -> curbuf(f32) + cur16(f16 for next layer) + this layer's output slice
        k_add_ln<<<LN_BLOCKS, 256, 0, stream>>>(curbuf, valbuf,
                                                ln2_g + (size_t)l * DM, ln2_b + (size_t)l * DM,
                                                curbuf, cur16, out + (size_t)l * LEN * DM);
    }
    (void)in_sizes; (void)n_in; (void)out_size; (void)ws_size;
}
